// MAGIC_LinkPredictor_12421045420439
// MI455X (gfx1250) — matrix operand order repaired and checked by reading, not compiled, not run on hardware
//
#include <hip/hip_runtime.h>
#include <hip/hip_bf16.h>
#include <stdint.h>

typedef __bf16 v16bf __attribute__((ext_vector_type(16)));
typedef float  v8f   __attribute__((ext_vector_type(8)));

#define WMMA_BF16(a, b, c) \
  __builtin_amdgcn_wmma_f32_16x16x32_bf16(false, (a), false, (b), (short)0, (c), false, false)

static __device__ __forceinline__ __bf16 f2bf(float x) { return (__bf16)x; }

// gfx1250 async copy: 16 bytes global -> LDS per lane, tracked on ASYNCcnt.
static __device__ __forceinline__ void async_copy16(uint32_t lds_byte_addr, const float* g) {
  asm volatile("global_load_async_to_lds_b128 %0, %1, off"
               :: "v"(lds_byte_addr), "v"((unsigned long long)(uintptr_t)g)
               : "memory");
}
static __device__ __forceinline__ void wait_async0() {
  asm volatile("s_wait_asynccnt 0x0" ::: "memory");
}

// A-fragment (16x32 bf16, M x K) built from f32 LDS row-major [16][lda] with on-the-fly cvt.
// ISA 7.12.2: lanes 0-15 hold row m=lane, K=0..7 (v0..3) + K=16..23 (v4..7);
// lanes 16-31 hold K=8..15 + K=24..31.
static __device__ __forceinline__ v16bf load_a_lds_f32(const float* sA, int lda, int koff, int lane) {
  const int m = lane & 15;
  const int kb = (lane >> 4) * 8;
  v16bf a;
#pragma unroll
  for (int j = 0; j < 8; ++j) {
    a[j]     = f2bf(sA[m * lda + koff + kb + j]);
    a[j + 8] = f2bf(sA[m * lda + koff + 16 + kb + j]);
  }
  return a;
}

// Same layout, LDS already bf16.
static __device__ __forceinline__ v16bf load_a_lds(const __bf16* sA, int lda, int koff, int lane) {
  const int m = lane & 15;
  const int kb = (lane >> 4) * 8;
  v16bf a;
#pragma unroll
  for (int j = 0; j < 8; ++j) {
    a[j]     = sA[m * lda + koff + kb + j];
    a[j + 8] = sA[m * lda + koff + 16 + kb + j];
  }
  return a;
}

// B-fragment (32x16 bf16, K x N) from global row-major W[K][ldw], cols n0..n0+15.
static __device__ __forceinline__ v16bf load_b_glb(const __bf16* W, int ldw, int koff, int n0, int lane) {
  const int n = lane & 15;
  const int kb = (lane >> 4) * 8;
  v16bf b;
#pragma unroll
  for (int j = 0; j < 8; ++j) {
    b[j]     = W[(size_t)(koff + kb + j) * ldw + n0 + n];
    b[j + 8] = W[(size_t)(koff + 16 + kb + j) * ldw + n0 + n];
  }
  return b;
}

__global__ void zero_f32(float* __restrict__ p, int n) {
  for (int i = blockIdx.x * blockDim.x + threadIdx.x; i < n; i += gridDim.x * blockDim.x)
    p[i] = 0.0f;
}

__global__ void cvt_bf16(const float* __restrict__ src, __bf16* __restrict__ dst, int n) {
  for (int i = blockIdx.x * blockDim.x + threadIdx.x; i < n; i += gridDim.x * blockDim.x)
    dst[i] = f2bf(src[i]);
}

// Fused: e = edge_attr @ anet_W + b ; msg = relu(h[src]+e) ; atomic scatter-add to agg[dst].
__global__ __launch_bounds__(256)
void edge_msg_scatter(const float* __restrict__ edge_attr,  // [E,64]
                      const int* __restrict__ src, const int* __restrict__ dst,
                      const float* __restrict__ h,          // [N,128]
                      const __bf16* __restrict__ Wbf,       // [64,128] bf16
                      const float* __restrict__ bias,       // [128]
                      float* __restrict__ agg,              // [N,128]
                      int E) {
  __shared__ float sAf[16 * 64];   // f32 tile staged by async DMA
  __shared__ int sSrc[16], sDst[16];
  const int tid = threadIdx.x;
  const int lane = tid & 31;
  const int wave = tid >> 5;
  const int n0 = wave * 16;
  const int nCol = n0 + (lane & 15);
  const int hi = lane >> 4;

  // Hoist B fragments (K=64 -> 2 chunks) and bias across the grid-stride loop.
  v16bf b0 = load_b_glb(Wbf, 128, 0, n0, lane);
  v16bf b1 = load_b_glb(Wbf, 128, 32, n0, lane);
  const float bn = bias[nCol];

  const uint32_t ldsDst = (uint32_t)(uintptr_t)&sAf[tid * 4];  // 16B per lane
  const size_t attrEnd = (size_t)E * 64 - 1;
  const int nTiles = (E + 15) >> 4;
  for (int tile = blockIdx.x; tile < nTiles; tile += gridDim.x) {
    const int e0 = tile << 4;
    const bool full = (e0 + 16 <= E);
    __syncthreads();
    if (tid < 16) {
      int e = e0 + tid;
      if (e >= E) e = E - 1;
      sSrc[tid] = src[e];
      sDst[tid] = dst[e];
    }
    if (full) {
      // Async DMA the contiguous 4KB tile into LDS (16B/lane, ASYNCcnt-tracked).
      async_copy16(ldsDst, edge_attr + (size_t)e0 * 64 + tid * 4);
      wait_async0();
    } else {
      // Generic tail: clamped scalar loads.
#pragma unroll
      for (int j = 0; j < 4; ++j) {
        int idx = tid * 4 + j;
        size_t gi = (size_t)e0 * 64 + idx;
        if (gi > attrEnd) gi = attrEnd;
        sAf[idx] = edge_attr[gi];
      }
    }
    // Prefetch the next tile this block will touch.
    {
      int nt = tile + gridDim.x;
      if (nt < nTiles)
        __builtin_prefetch(edge_attr + ((size_t)nt << 4) * 64 + tid * 4, 0, 1);
    }
    __syncthreads();

    v16bf a0 = load_a_lds_f32(sAf, 64, 0, lane);
    v16bf a1 = load_a_lds_f32(sAf, 64, 32, lane);
    v8f c = {};
    c = WMMA_BF16(a0, b0, c);
    c = WMMA_BF16(a1, b1, c);

    if (full) {
      // Straight-line epilogue: gather + relu + L2-resident atomic scatter.
#pragma unroll
      for (int r = 0; r < 8; ++r) {
        const int m = r + hi * 8;
        float v = c[r] + bn + h[(size_t)sSrc[m] * 128 + nCol];
        v = v > 0.0f ? v : 0.0f;
        atomicAdd(&agg[(size_t)sDst[m] * 128 + nCol], v);
      }
    } else {
#pragma unroll
      for (int r = 0; r < 8; ++r) {
        const int m = r + hi * 8;
        if (e0 + m < E) {
          float v = c[r] + bn + h[(size_t)sSrc[m] * 128 + nCol];
          v = v > 0.0f ? v : 0.0f;
          atomicAdd(&agg[(size_t)sDst[m] * 128 + nCol], v);
        }
      }
    }
  }
}

// z = concat([agg, (1+eps)*h]) @ mlp_W + b ; out = fuse_relu ? relu(z) : leaky(z)
__global__ __launch_bounds__(256)
void node_update(const float* __restrict__ agg, const float* __restrict__ h,
                 const __bf16* __restrict__ Wbf,  // [256,128] bf16
                 const float* __restrict__ bias, const float* __restrict__ epsp,
                 float* __restrict__ out, int N, int fuse_relu) {
  __shared__ __bf16 sA[16 * 256];
  const int tid = threadIdx.x;
  const int lane = tid & 31;
  const int wave = tid >> 5;
  const int n0 = wave * 16;
  const int nCol = n0 + (lane & 15);
  const int hi = lane >> 4;
  const float scale = 1.0f + epsp[0];
  const float bn = bias[nCol];

  // Hoist all 8 B K-chunks in registers.
  v16bf bf[8];
#pragma unroll
  for (int kc = 0; kc < 8; ++kc) bf[kc] = load_b_glb(Wbf, 128, kc * 32, n0, lane);

  const int nTiles = (N + 15) >> 4;
  for (int tile = blockIdx.x; tile < nTiles; tile += gridDim.x) {
    const int v0 = tile << 4;
    __syncthreads();
    // 16 rows x 256 K = 4096 elems, 16/thread.
#pragma unroll
    for (int j = 0; j < 16; ++j) {
      int idx = tid * 16 + j;
      int row = idx >> 8;
      int k = idx & 255;
      int node = v0 + row;
      if (node >= N) node = N - 1;
      float v = (k < 128) ? agg[(size_t)node * 128 + k]
                          : scale * h[(size_t)node * 128 + (k - 128)];
      sA[idx] = f2bf(v);
    }
    __syncthreads();

    v8f c = {};
#pragma unroll
    for (int kc = 0; kc < 8; ++kc) {
      v16bf a = load_a_lds(sA, 256, kc * 32, lane);
      c = WMMA_BF16(a, bf[kc], c);
    }

#pragma unroll
    for (int r = 0; r < 8; ++r) {
      const int m = r + hi * 8;
      const int node = v0 + m;
      if (node < N) {
        float z = c[r] + bn;
        float o = fuse_relu ? (z > 0.0f ? z : 0.0f)          // relu(leaky(z)) == relu(z)
                            : (z > 0.0f ? z : 0.2f * z);     // leaky
        out[(size_t)node * 128 + nCol] = o;
      }
    }
  }
}

// h_edge = h[src]*h[dst] (written f32) ; z = leaky(h_edge@W1+b1) ; yhat = softmax(z@W2+b2)
__global__ __launch_bounds__(256)
void link_predict(const int* __restrict__ src, const int* __restrict__ dst,
                  const float* __restrict__ h,      // [N,128] post-relu
                  const __bf16* __restrict__ W1bf,  // [128,128] bf16
                  const float* __restrict__ b1,
                  const float* __restrict__ W2,     // [128,2]
                  const float* __restrict__ b2,     // [2]
                  float* __restrict__ hedge_out,    // [E,128]
                  float* __restrict__ yhat_out,     // [E,2]
                  int E) {
  __shared__ __bf16 sHE[16 * 128];
  __shared__ float sZ[16 * 128];
  __shared__ int sSrc[16], sDst[16];
  const int tid = threadIdx.x;
  const int lane = tid & 31;
  const int wave = tid >> 5;
  const int n0 = wave * 16;
  const int nCol = n0 + (lane & 15);
  const int hi = lane >> 4;
  const float bn = b1[nCol];

  v16bf bf[4];
#pragma unroll
  for (int kc = 0; kc < 4; ++kc) bf[kc] = load_b_glb(W1bf, 128, kc * 32, n0, lane);

  const int nTiles = (E + 15) >> 4;
  for (int tile = blockIdx.x; tile < nTiles; tile += gridDim.x) {
    const int e0 = tile << 4;
    const bool full = (e0 + 16 <= E);
    __syncthreads();
    if (tid < 16) {
      int e = e0 + tid;
      if (e >= E) e = E - 1;
      sSrc[tid] = src[e];
      sDst[tid] = dst[e];
    }
    __syncthreads();
    // h_edge tile: 16x128 = 2048 elems, 8/thread. Gathers hit L2 (h is 25.6MB).
#pragma unroll
    for (int j = 0; j < 8; ++j) {
      int idx = tid * 8 + j;
      int row = idx >> 7;
      int k = idx & 127;
      float v = h[(size_t)sSrc[row] * 128 + k] * h[(size_t)sDst[row] * 128 + k];
      sHE[idx] = f2bf(v);
      if (full || (e0 + row < E)) hedge_out[(size_t)(e0 + row) * 128 + k] = v;
    }
    __syncthreads();

    v8f c = {};
#pragma unroll
    for (int kc = 0; kc < 4; ++kc) {
      v16bf a = load_a_lds(sHE, 128, kc * 32, lane);
      c = WMMA_BF16(a, bf[kc], c);
    }
#pragma unroll
    for (int r = 0; r < 8; ++r) {
      const int m = r + hi * 8;
      float z = c[r] + bn;
      sZ[m * 128 + nCol] = z > 0.0f ? z : 0.2f * z;  // leaky
    }
    __syncthreads();

    if (tid < 16) {
      const int e = e0 + tid;
      if (e < E) {
        float a0 = b2[0], a1 = b2[1];
        const float* zr = &sZ[tid * 128];
#pragma unroll 4
        for (int k = 0; k < 128; ++k) {
          float zv = zr[k];
          a0 += zv * W2[k * 2 + 0];
          a1 += zv * W2[k * 2 + 1];
        }
        float mx = a0 > a1 ? a0 : a1;
        float x0 = __expf(a0 - mx), x1 = __expf(a1 - mx);
        float s = x0 + x1;
        yhat_out[(size_t)e * 2 + 0] = x0 / s;
        yhat_out[(size_t)e * 2 + 1] = x1 / s;
      }
    }
  }
}

extern "C" void kernel_launch(void* const* d_in, const int* in_sizes, int n_in,
                              void* d_out, int out_size, void* d_ws, size_t ws_size,
                              hipStream_t stream) {
  const int N = 50000, E = 800000;
  const float* x        = (const float*)d_in[0];
  const int*   ei       = (const int*)d_in[1];   // [2,E]
  const float* edgeattr = (const float*)d_in[2];
  const float* anet_W0  = (const float*)d_in[3];
  const float* anet_b0  = (const float*)d_in[4];
  const float* mlp_W0   = (const float*)d_in[5];
  const float* mlp_b0   = (const float*)d_in[6];
  const float* eps0     = (const float*)d_in[7];
  const float* anet_W1  = (const float*)d_in[8];
  const float* anet_b1  = (const float*)d_in[9];
  const float* mlp_W1   = (const float*)d_in[10];
  const float* mlp_b1   = (const float*)d_in[11];
  const float* eps1     = (const float*)d_in[12];
  const float* lp_W1    = (const float*)d_in[13];
  const float* lp_b1    = (const float*)d_in[14];
  const float* lp_W2    = (const float*)d_in[15];
  const float* lp_b2    = (const float*)d_in[16];
  const int* src = ei;
  const int* dst = ei + E;

  // Workspace layout: agg | h1 | h2 | bf16 weights
  float* agg = (float*)d_ws;
  float* h1  = agg + (size_t)N * 128;
  float* h2  = h1 + (size_t)N * 128;
  __bf16* aW0 = (__bf16*)(h2 + (size_t)N * 128);
  __bf16* aW1 = aW0 + 64 * 128;
  __bf16* mW0 = aW1 + 64 * 128;
  __bf16* mW1 = mW0 + 256 * 128;
  __bf16* lW1 = mW1 + 256 * 128;

  // Pre-convert weights to bf16 (tiny, once per launch).
  cvt_bf16<<<32, 256, 0, stream>>>(anet_W0, aW0, 64 * 128);
  cvt_bf16<<<32, 256, 0, stream>>>(anet_W1, aW1, 64 * 128);
  cvt_bf16<<<128, 256, 0, stream>>>(mlp_W0, mW0, 256 * 128);
  cvt_bf16<<<128, 256, 0, stream>>>(mlp_W1, mW1, 256 * 128);
  cvt_bf16<<<64, 256, 0, stream>>>(lp_W1, lW1, 128 * 128);

  // Layer 0
  zero_f32<<<2048, 256, 0, stream>>>(agg, N * 128);
  edge_msg_scatter<<<3200, 256, 0, stream>>>(edgeattr, src, dst, x, aW0, anet_b0, agg, E);
  node_update<<<1024, 256, 0, stream>>>(agg, x, mW0, mlp_b0, eps0, h1, N, /*fuse_relu=*/0);

  // Layer 1 (fuses the final relu: relu(leaky(z)) == relu(z))
  zero_f32<<<2048, 256, 0, stream>>>(agg, N * 128);
  edge_msg_scatter<<<3200, 256, 0, stream>>>(edgeattr, src, dst, h1, aW1, anet_b1, agg, E);
  node_update<<<1024, 256, 0, stream>>>(agg, h1, mW1, mlp_b1, eps1, h2, N, /*fuse_relu=*/1);

  // Link predictor -> d_out = [h_edge (E*128) | yhat (E*2)]
  float* hedge_out = (float*)d_out;
  float* yhat_out  = hedge_out + (size_t)E * 128;
  link_predict<<<3200, 256, 0, stream>>>(src, dst, h2, lW1, lp_b1, lp_W2, lp_b2,
                                         hedge_out, yhat_out, E);
}
